// Pure1BitLinear_24180665876586
// MI455X (gfx1250) — 8-bit matrix operand order repaired and checked by reading, compiled, not run on hardware
//
#include <hip/hip_runtime.h>


typedef __attribute__((ext_vector_type(8)))  int   v8i;
typedef __attribute__((ext_vector_type(4)))  int   v4i;
typedef __attribute__((ext_vector_type(4)))  float v4f;

#define M_TOK 32768
#define K_DIM 2048
#define N_DIM 2048
#define KTILES (K_DIM / 64)

__global__ __launch_bounds__(256) void wconv_kernel(const float* __restrict__ W,
                                                    signed char* __restrict__ W8) {
  const int total = (N_DIM * K_DIM) / 4;
  for (int idx = blockIdx.x * blockDim.x + threadIdx.x; idx < total;
       idx += gridDim.x * blockDim.x) {
    v4f w = __builtin_nontemporal_load((const v4f*)W + idx);
    int q0 = (int)w.x, q1 = (int)w.y, q2 = (int)w.z, q3 = (int)w.w;
    ((int*)W8)[idx] = (q0 & 255) | ((q1 & 255) << 8) |
                      ((q2 & 255) << 16) | ((q3 & 255) << 24);
  }
}

__global__ __launch_bounds__(256) void quant_kernel(const float* __restrict__ x,
                                                    signed char* __restrict__ xq,
                                                    float* __restrict__ scales) {
  const int lane = threadIdx.x & 31;
  const int row  = blockIdx.x * 8 + (threadIdx.x >> 5);
  const float* xr = x + (size_t)row * K_DIM;

  v4f v[16];
  float amax = 0.0f;
#pragma unroll
  for (int j = 0; j < 16; ++j) {
    v[j] = __builtin_nontemporal_load((const v4f*)xr + lane + 32 * j);
    amax = fmaxf(amax, fmaxf(fmaxf(fabsf(v[j].x), fabsf(v[j].y)),
                             fmaxf(fabsf(v[j].z), fabsf(v[j].w))));
  }
#pragma unroll
  for (int off = 16; off; off >>= 1)
    amax = fmaxf(amax, __shfl_xor(amax, off, 32));
  amax = fmaxf(amax, 1e-5f);
  const float s = 127.0f / amax;
  if (lane == 0) scales[row] = amax * (1.0f / 127.0f);

  signed char* tbase = xq + (size_t)(row >> 4) * KTILES * 1024;
  const int mlo = row & 15;
#pragma unroll
  for (int j = 0; j < 16; ++j) {
    int q0 = (int)rintf(fminf(fmaxf(v[j].x * s, -128.0f), 127.0f));
    int q1 = (int)rintf(fminf(fmaxf(v[j].y * s, -128.0f), 127.0f));
    int q2 = (int)rintf(fminf(fmaxf(v[j].z * s, -128.0f), 127.0f));
    int q3 = (int)rintf(fminf(fmaxf(v[j].w * s, -128.0f), 127.0f));
    int pk = (q0 & 255) | ((q1 & 255) << 8) | ((q2 & 255) << 16) |
             ((q3 & 255) << 24);
    const int k     = 4 * (lane + 32 * j);
    const int kt    = k >> 6;
    const int k_off = k & 63;
    const int h     = (k_off >> 4) & 1;
    const int t     = k_off - 16 * h;
    const int vslot = ((t >> 2) & 3) + ((t >> 5) << 2);
    *(int*)(tbase + (size_t)kt * 1024 + (mlo + 16 * h) * 32 + vslot * 4) = pk;
  }
}

__global__ __launch_bounds__(256) void gemm_kernel(
    const signed char* __restrict__ Xq, const signed char* __restrict__ W8,
    const float* __restrict__ scales, float* __restrict__ out) {
  const int lane = threadIdx.x & 31;
  const int wave = threadIdx.x >> 5;
  const int wm = wave >> 1;
  const int wn = wave & 1;
  const int l15 = lane & 15;
  const int hi  = lane >> 4;

  const int m0 = blockIdx.x * 128 + wm * 32;
  const int n0 = blockIdx.y * 128 + wn * 64;

  const signed char* abase0 =
      Xq + ((size_t)(m0 >> 4) * KTILES) * 1024 + lane * 32;
  const signed char* abase1 = abase0 + (size_t)KTILES * 1024;
  const signed char* bbase = W8 + (size_t)(n0 + l15) * K_DIM + hi * 16;

  v8i acc[2][4] = {};

  for (int kt = 0; kt < KTILES; ++kt) {
    v8i A[2], B[4];
#pragma unroll
    for (int mt = 0; mt < 2; ++mt) {
      const signed char* p = (mt ? abase1 : abase0) + (size_t)kt * 1024;
      v4i lo = *(const v4i*)(p + 0);
      v4i hi4 = *(const v4i*)(p + 16);
      v8i t = {lo.x, lo.y, lo.z, lo.w, hi4.x, hi4.y, hi4.z, hi4.w};
      A[mt] = t;
    }
#pragma unroll
    for (int nt = 0; nt < 4; ++nt) {
      const signed char* p = bbase + (size_t)nt * 16 * K_DIM + kt * 64;
      v4i blo = *(const v4i*)(p + 0);
      v4i bhi = *(const v4i*)(p + 32);
      v8i t = {blo.x, blo.y, blo.z, blo.w, bhi.x, bhi.y, bhi.z, bhi.w};
      B[nt] = t;
    }
#pragma unroll
    for (int mt = 0; mt < 2; ++mt)
#pragma unroll
      for (int nt = 0; nt < 4; ++nt)
        acc[mt][nt] = __builtin_amdgcn_wmma_i32_16x16x64_iu8(
             true, A[mt],  true, B[nt], acc[mt][nt],
             false,  false);
  }

#pragma unroll
  for (int mt = 0; mt < 2; ++mt) {
#pragma unroll
    for (int r = 0; r < 8; ++r) {
      const int m = m0 + mt * 16 + hi * 8 + r;
      const float s = scales[m];
#pragma unroll
      for (int nt = 0; nt < 4; ++nt) {
        const int n = n0 + nt * 16 + l15;
        __builtin_nontemporal_store((float)acc[mt][nt][r] * s,
                                    out + (size_t)m * N_DIM + n);
      }
    }
  }
}

extern "C" void kernel_launch(void* const* d_in, const int* in_sizes, int n_in,
                              void* d_out, int out_size, void* d_ws, size_t ws_size,
                              hipStream_t stream) {
  const float* x = (const float*)d_in[0];
  const float* W = (const float*)d_in[1];
  float* out = (float*)d_out;

  char* ws = (char*)d_ws;
  signed char* Xq = (signed char*)ws;
  signed char* W8 = (signed char*)(ws + (size_t)M_TOK * K_DIM);
  float* scales   = (float*)(ws + (size_t)M_TOK * K_DIM + (size_t)N_DIM * K_DIM);

  wconv_kernel<<<1024, 256, 0, stream>>>(W, W8);
  quant_kernel<<<M_TOK / 8, 256, 0, stream>>>(x, Xq, scales);

  dim3 grid(M_TOK / 128, N_DIM / 128);
  gemm_kernel<<<grid, 256, 0, stream>>>(Xq, W8, scales, out);
}
